// CondInst_89850715832827
// MI455X (gfx1250) — matrix operand order repaired and checked by reading, not compiled, not run on hardware
//
#include <hip/hip_runtime.h>
#include <hip/hip_bf16.h>

#define Hd 160
#define Wd 160
#define HW (Hd*Wd)
#define CI 128
#define KTILES 36            // K = 9 taps * 128 ci = 1152 = 36 * 32
#define ROWS 4               // output rows per workgroup (weight reuse x4)
#define LAYER_WELEMS (8*KTILES*32*16)   // per-layer packed weight elements (bf16)

typedef __bf16 bf16_t;
typedef __attribute__((ext_vector_type(16))) __bf16 v16bf;
typedef __attribute__((ext_vector_type(8)))  float  v8f;

// ---- CDNA5 async global<->LDS path (guarded; sync fallback otherwise) ----
#if defined(__has_builtin)
#  if __has_builtin(__builtin_amdgcn_global_load_async_to_lds_b128) && \
      __has_builtin(__builtin_amdgcn_global_store_async_from_lds_b128) && \
      __has_builtin(__builtin_amdgcn_s_wait_asynccnt)
#    define HAVE_ASYNC 1
#  endif
#endif
#ifndef HAVE_ASYNC
#  define HAVE_ASYNC 0
#endif

#if HAVE_ASYNC
typedef int v4i_vec __attribute__((vector_size(16)));
typedef __attribute__((address_space(1))) v4i_vec g_v4i;   // global int4
typedef __attribute__((address_space(3))) v4i_vec l_v4i;   // LDS int4
#endif

// ---------------------------------------------------------------------------
// Pack tower weights into WMMA B-matrix per-lane layout (bf16).
// B is K(32) x N(16): lane l holds column N=l%16, element j is K=8h+j (j<8) or 16+8h+(j-8), h=l/16.
// wpack layout: [layer][ntile(8)][ktile(36)][lane(32)][16]
// ---------------------------------------------------------------------------
__global__ __launch_bounds__(256) void pack_weights(const float* __restrict__ tw,
                                                    bf16_t* __restrict__ wpack) {
    int idx = blockIdx.x * 256 + threadIdx.x;
    if (idx >= 4 * 8 * KTILES * 32) return;
    int lane  = idx & 31;
    int kt    = (idx >> 5) % KTILES;
    int nt    = ((idx >> 5) / KTILES) % 8;
    int layer = idx / (32 * KTILES * 8);
    int co  = nt * 16 + (lane & 15);
    int hh  = lane >> 4;
    bf16_t* dst = wpack + ((((size_t)layer * 8 + nt) * KTILES + kt) * 32 + lane) * 16;
#pragma unroll
    for (int j = 0; j < 16; ++j) {
        int k   = kt * 32 + ((j < 8) ? (hh * 8 + j) : (16 + hh * 8 + (j - 8)));
        int tap = k >> 7;
        int ci  = k & 127;
        int ky  = tap / 3, kx = tap % 3;
        float w = tw[((((size_t)layer * 128 + co) * 128 + ci) * 3 + ky) * 3 + kx];
        dst[j] = (bf16_t)w;
    }
}

// Fold BN into per-channel scale/shift:  y = relu(x*inv + shift)
__global__ void bn_prep(const float* __restrict__ g, const float* __restrict__ b,
                        const float* __restrict__ m, const float* __restrict__ v,
                        float* __restrict__ inv, float* __restrict__ sh) {
    int i = blockIdx.x * 128 + threadIdx.x;   // 4 layers * 128 ch = 512
    if (i >= 512) return;
    float iv = g[i] * rsqrtf(v[i] + 1e-5f);
    inv[i] = iv;
    sh[i]  = b[i] - m[i] * iv;
}

// cnn_feature batch 0: NCHW f32 -> HWC bf16 (channels-last, WMMA-friendly)
__global__ __launch_bounds__(256) void to_bf16_hwc(const float* __restrict__ x,
                                                   bf16_t* __restrict__ xo) {
    int p = blockIdx.x * 256 + threadIdx.x;
    if (p >= HW) return;
    alignas(16) bf16_t tmp[CI];
#pragma unroll 8
    for (int c = 0; c < CI; ++c) tmp[c] = (bf16_t)x[(size_t)c * HW + p];
    uint4* dst = (uint4*)(xo + (size_t)p * CI);
    const uint4* src = (const uint4*)tmp;
#pragma unroll
    for (int j = 0; j < 16; ++j) dst[j] = src[j];
}

// ---------------------------------------------------------------------------
// conv3x3(128->128, SAME) + BN + ReLU via v_wmma_f32_16x16x32_bf16.
// Workgroup tile: 4 rows x 32 px x 128 co. k-tile loop is outermost so each
// per-lane B fragment (weights) feeds 8 WMMAs (4 rows x 2 pixel-halves).
// Halo staged with CDNA5 async global->LDS; epilogue stored async from LDS.
// ---------------------------------------------------------------------------
__global__ __launch_bounds__(256) void conv3x3_bn_relu(const bf16_t* __restrict__ in,    // [HW][128]
                                                       const bf16_t* __restrict__ wpack, // this layer
                                                       const float* __restrict__ bninv,  // [128]
                                                       const float* __restrict__ bnsh,   // [128]
                                                       bf16_t* __restrict__ out) {       // [HW][128]
    __shared__ alignas(16) bf16_t sIn[(ROWS + 2) * 34 * CI];   // rows y0-1..y0+4, x0-1..x0+32
    __shared__ alignas(16) bf16_t sOut[ROWS * 32 * CI];

    const int tid  = threadIdx.x;
    const int lane = tid & 31;
    const int wave = tid >> 5;         // = co ntile 0..7
    const int x0   = blockIdx.x * 32;
    const int y0   = blockIdx.y * ROWS;
    const int sub  = tid & 7;          // 32B chunk within a pixel's 128 channels

    // Stage halo tile: (ROWS+2)*34 pixel-rows * 256B, 8 threads per pixel-row.
    for (int pr = (tid >> 3); pr < (ROWS + 2) * 34; pr += 32) {
        int dy = pr / 34, xi = pr % 34;
        int gy = y0 + dy - 1, gx = x0 + xi - 1;
        bf16_t* ldst = sIn + (size_t)pr * CI + sub * 16;
        if (gy >= 0 && gy < Hd && gx >= 0 && gx < Wd) {
            const bf16_t* gsrc = in + ((size_t)(gy * Wd + gx)) * CI + sub * 16;
#if HAVE_ASYNC
            __builtin_amdgcn_global_load_async_to_lds_b128((g_v4i*)gsrc, (l_v4i*)ldst, 0, 0);
            __builtin_amdgcn_global_load_async_to_lds_b128((g_v4i*)gsrc, (l_v4i*)ldst, 16, 0);
#else
            const uint4* s = (const uint4*)gsrc;
            uint4* d = (uint4*)ldst;
            d[0] = s[0]; d[1] = s[1];
#endif
        } else {
            uint4 z = make_uint4(0u, 0u, 0u, 0u);
            uint4* d = (uint4*)ldst;
            d[0] = z; d[1] = z;
        }
    }
#if HAVE_ASYNC
    __builtin_amdgcn_s_wait_asynccnt(0);
#endif
    __syncthreads();

    const int mrow = lane & 15;            // A-matrix M row held by this lane
    const int csel = (lane < 16) ? 0 : 8;  // element offset of this lane's K-halves
    v8f acc[ROWS][2];
#pragma unroll
    for (int r = 0; r < ROWS; ++r) { acc[r][0] = v8f{}; acc[r][1] = v8f{}; }

    const bf16_t* wp = wpack + ((size_t)wave * KTILES) * 32 * 16 + (size_t)lane * 16;

#pragma unroll 2
    for (int kt = 0; kt < KTILES; ++kt) {
        int tap = kt >> 2;                     // 0..8
        int ky = tap / 3, kx = tap % 3;
        int cbase = (kt & 3) * 32 + csel;      // lane<16 -> K {0..7,16..23}; else {8..15,24..31}
        // B fragment: prepacked weights, 32 contiguous bytes per lane
        union { v16bf v; uint4 q[2]; } bf_;
        const uint4* bsrc = (const uint4*)(wp + (size_t)kt * 32 * 16);
        bf_.q[0] = bsrc[0]; bf_.q[1] = bsrc[1];
#pragma unroll
        for (int r = 0; r < ROWS; ++r) {
            int rb0 = ((r + ky) * 34 + (mrow + kx)) * CI + cbase;        // pixels 0..15
            int rb1 = ((r + ky) * 34 + (16 + mrow + kx)) * CI + cbase;   // pixels 16..31
            union { v16bf v; uint4 q[2]; } a0, a1;
            a0.q[0] = *(const uint4*)(sIn + rb0);
            a0.q[1] = *(const uint4*)(sIn + rb0 + 16);
            a1.q[0] = *(const uint4*)(sIn + rb1);
            a1.q[1] = *(const uint4*)(sIn + rb1 + 16);
            acc[r][0] = __builtin_amdgcn_wmma_f32_16x16x32_bf16(false, a0.v, false, bf_.v,
                                                                (short)0, acc[r][0], false, false);
            acc[r][1] = __builtin_amdgcn_wmma_f32_16x16x32_bf16(false, a1.v, false, bf_.v,
                                                                (short)0, acc[r][1], false, false);
        }
    }

    // BN + ReLU, transpose through LDS for coalesced channels-last stores.
    const int co = wave * 16 + mrow;           // D: N = lane%16
    const float inv = bninv[co], sh = bnsh[co];
    const int phalf = (lane >> 4) << 3;        // D: M = v + 8*(lane>=16)
#pragma unroll
    for (int r = 0; r < ROWS; ++r) {
#pragma unroll
        for (int v = 0; v < 8; ++v) {
            int px = v + phalf;
            sOut[(r * 32 + px) * CI + co]      = (bf16_t)fmaxf(acc[r][0][v] * inv + sh, 0.f);
            sOut[(r * 32 + 16 + px) * CI + co] = (bf16_t)fmaxf(acc[r][1][v] * inv + sh, 0.f);
        }
    }
    __syncthreads();
    for (int pr = (tid >> 3); pr < ROWS * 32; pr += 32) {
        int r = pr >> 5, px = pr & 31;
        bf16_t* lsrc = sOut + (size_t)pr * CI + sub * 16;
        bf16_t* gdst = out + ((size_t)((y0 + r) * Wd + x0 + px)) * CI + sub * 16;
#if HAVE_ASYNC
        __builtin_amdgcn_global_store_async_from_lds_b128((g_v4i*)gdst, (l_v4i*)lsrc, 0, 0);
        __builtin_amdgcn_global_store_async_from_lds_b128((g_v4i*)gdst, (l_v4i*)lsrc, 16, 0);
#else
        const uint4* s = (const uint4*)lsrc;
        uint4* d = (uint4*)gdst;
        d[0] = s[0]; d[1] = s[1];
#endif
    }
    // S_ENDPGM performs an implicit wait-idle, covering outstanding async stores.
}

// Controller conv evaluated only at the 100 detection points (169 outs each).
__global__ __launch_bounds__(256) void controller(const float* __restrict__ x,   // batch0 NCHW
                                                  const float* __restrict__ cw,  // [169][128][3][3]
                                                  const float* __restrict__ cb,
                                                  const int* __restrict__ det,
                                                  float* __restrict__ params) {  // [100][169]
    __shared__ float patch[CI * 9];
    const int k = blockIdx.x;
    const int xs = det[2 * k], ys = det[2 * k + 1];
    const int tid = threadIdx.x;
    for (int i = tid; i < CI * 9; i += 256) {
        int ci = i / 9, r = i % 9;
        int dy = r / 3, dx = r % 3;
        int gy = ys + dy - 1, gx = xs + dx - 1;
        float v = 0.f;
        if (gy >= 0 && gy < Hd && gx >= 0 && gx < Wd) v = x[(size_t)ci * HW + gy * Wd + gx];
        patch[i] = v;
    }
    __syncthreads();
    for (int o = tid; o < 169; o += 256) {
        float acc = cb[o];
        const float* wrow = cw + (size_t)o * (CI * 9);
        for (int i = 0; i < CI * 9; ++i) acc += wrow[i] * patch[i];
        params[k * 169 + o] = acc;
    }
}

// mask_feats[0]: 128 -> 8 projection, channels-last bf16 input.
__global__ __launch_bounds__(256) void projection(const bf16_t* __restrict__ x4,
                                                  const float* __restrict__ pw,  // [8][128]
                                                  const float* __restrict__ pb,
                                                  float* __restrict__ feats) {   // [HW][8]
    __shared__ float sw[8 * CI];
    __shared__ float sb[8];
    const int tid = threadIdx.x;
    for (int i = tid; i < 8 * CI; i += 256) sw[i] = pw[i];
    if (tid < 8) sb[tid] = pb[tid];
    __syncthreads();
    int p = blockIdx.x * 256 + tid;
    if (p >= HW) return;
    float acc[8];
#pragma unroll
    for (int o = 0; o < 8; ++o) acc[o] = sb[o];
    const bf16_t* xr = x4 + (size_t)p * CI;
    for (int c = 0; c < CI; ++c) {
        float xv = (float)xr[c];
#pragma unroll
        for (int o = 0; o < 8; ++o) acc[o] += xv * sw[o * CI + c];
    }
#pragma unroll
    for (int o = 0; o < 8; ++o) feats[(size_t)p * 8 + o] = acc[o];
}

// Dynamic per-detection MLP: 10 -> 8 -> 8 -> 1 over all pixels.
__global__ __launch_bounds__(256) void mlp(const float* __restrict__ params,
                                           const float* __restrict__ feats,
                                           const int* __restrict__ det,
                                           float* __restrict__ outp) {
    __shared__ float sp[169];
    const int k = blockIdx.y;
    const int tid = threadIdx.x;
    if (tid < 169) sp[tid] = params[k * 169 + tid];
    __syncthreads();
    int p = blockIdx.x * 256 + tid;
    if (p >= HW) return;
    int py = p / Wd, px = p % Wd;
    float h[10];
    h[0] = (float)(det[2 * k] * 4) - (float)(px * 4 + 2);
    h[1] = (float)(det[2 * k + 1] * 4) - (float)(py * 4 + 2);
    const float* f = feats + (size_t)p * 8;
#pragma unroll
    for (int c = 0; c < 8; ++c) h[2 + c] = f[c];
    float h1[8], h2[8];
#pragma unroll
    for (int o = 0; o < 8; ++o) {
        float s = sp[152 + o];
#pragma unroll
        for (int c = 0; c < 10; ++c) s += sp[o * 10 + c] * h[c];
        h1[o] = fmaxf(s, 0.f);
    }
#pragma unroll
    for (int o = 0; o < 8; ++o) {
        float s = sp[160 + o];
#pragma unroll
        for (int c = 0; c < 8; ++c) s += sp[80 + o * 8 + c] * h1[c];
        h2[o] = fmaxf(s, 0.f);
    }
    float s = sp[168];
#pragma unroll
    for (int c = 0; c < 8; ++c) s += sp[144 + c] * h2[c];
    outp[(size_t)k * HW + p] = s;
}

extern "C" void kernel_launch(void* const* d_in, const int* in_sizes, int n_in,
                              void* d_out, int out_size, void* d_ws, size_t ws_size,
                              hipStream_t stream) {
    const float* cnn = (const float*)d_in[0];   // [4][128][160][160]; only batch 0 matters
    const float* tw  = (const float*)d_in[1];   // [4][128][128][3][3]
    const float* g   = (const float*)d_in[2];
    const float* be  = (const float*)d_in[3];
    const float* mn  = (const float*)d_in[4];
    const float* vr  = (const float*)d_in[5];
    const float* pw  = (const float*)d_in[6];
    const float* pb  = (const float*)d_in[7];
    const float* cw  = (const float*)d_in[8];
    const float* cb  = (const float*)d_in[9];
    const int*   det = (const int*)d_in[10];
    float* outp = (float*)d_out;

    char* ws = (char*)d_ws;
    size_t off = 0;
    auto alloc = [&](size_t bytes) { size_t o = off; off = (off + bytes + 255) & ~(size_t)255; return o; };
    bf16_t* xin   = (bf16_t*)(ws + alloc((size_t)HW * CI * 2));
    bf16_t* buf1  = (bf16_t*)(ws + alloc((size_t)HW * CI * 2));
    bf16_t* buf2  = (bf16_t*)(ws + alloc((size_t)HW * CI * 2));
    bf16_t* wpk   = (bf16_t*)(ws + alloc((size_t)4 * LAYER_WELEMS * 2));
    float*  bninv = (float*) (ws + alloc(512 * 4));
    float*  bnsh  = (float*) (ws + alloc(512 * 4));
    float*  feats = (float*) (ws + alloc((size_t)HW * 8 * 4));
    float*  prm   = (float*) (ws + alloc(100 * 169 * 4));

    pack_weights<<<(4 * 8 * KTILES * 32 + 255) / 256, 256, 0, stream>>>(tw, wpk);
    bn_prep<<<4, 128, 0, stream>>>(g, be, mn, vr, bninv, bnsh);
    to_bf16_hwc<<<HW / 256, 256, 0, stream>>>(cnn, xin);

    dim3 cg(Wd / 32, Hd / ROWS);
    conv3x3_bn_relu<<<cg, 256, 0, stream>>>(xin,  wpk + 0 * (size_t)LAYER_WELEMS, bninv + 0,   bnsh + 0,   buf1);
    conv3x3_bn_relu<<<cg, 256, 0, stream>>>(buf1, wpk + 1 * (size_t)LAYER_WELEMS, bninv + 128, bnsh + 128, buf2);
    conv3x3_bn_relu<<<cg, 256, 0, stream>>>(buf2, wpk + 2 * (size_t)LAYER_WELEMS, bninv + 256, bnsh + 256, buf1);
    conv3x3_bn_relu<<<cg, 256, 0, stream>>>(buf1, wpk + 3 * (size_t)LAYER_WELEMS, bninv + 384, bnsh + 384, buf2);

    controller<<<100, 256, 0, stream>>>(cnn, cw, cb, det, prm);
    projection<<<HW / 256, 256, 0, stream>>>(buf2, pw, pb, feats);
    mlp<<<dim3(HW / 256, 100), 256, 0, stream>>>(prm, feats, det, outp);
}
